// DeformableConv3D_82678120448138
// MI455X (gfx1250) — matrix operand order repaired and checked by reading, not compiled, not run on hardware
//
#include <hip/hip_runtime.h>
#include <hip/hip_bf16.h>

// ---- CDNA5 WMMA vector types (wave32) ----
typedef __attribute__((ext_vector_type(16))) _Float16 v16h;
typedef __attribute__((ext_vector_type(8)))  _Float16 v8h;
typedef __attribute__((ext_vector_type(8)))  float    v8f;

#define CIN_     32
#define COUT_    32
#define KTAPS    27          // 3x3x3
#define DD       32
#define HH       64
#define WW       64
#define NSPATIAL (DD*HH*WW)  // 131072
#define PD       34          // D+2
#define PH       66
#define PW       66
#define XPAD_ELEMS (PD*PH*PW*CIN_)          // 4,739,328 f16
#define W1F_ELEMS  (KTAPS*7*32*16)          // 96,768  f16 (108->7 n-tiles of 16, padded)
#define W2F_ELEMS  (KTAPS*2*32*16)          // 27,648  f16

// ---------------------------------------------------------------------------
// Kernel 1: restage x (NCDHW f32) -> zero-padded channels-last f16
//           xpad[zp][yp][xp][cin], zp in [0,34), halo = 1
// ---------------------------------------------------------------------------
__global__ __launch_bounds__(256) void k_pad_x(const float* __restrict__ x,
                                               _Float16* __restrict__ xpad) {
    int idx = blockIdx.x * 256 + threadIdx.x;
    if (idx >= XPAD_ELEMS) return;
    int cin = idx & 31;
    int t   = idx >> 5;
    int xp  = t % PW; t /= PW;
    int yp  = t % PH;
    int zp  = t / PH;
    float v = 0.f;
    if (zp >= 1 && zp <= DD && yp >= 1 && yp <= HH && xp >= 1 && xp <= WW)
        v = x[((cin * DD + (zp - 1)) * HH + (yp - 1)) * WW + (xp - 1)];
    xpad[idx] = (_Float16)v;
}

// ---------------------------------------------------------------------------
// Kernel 2: pack weights into per-lane WMMA B-fragment layout.
//   B frag (K=32 x N=16, f16): lane<16 -> n=lane,   elems e = K 0..7, 16..23
//                              lane>=16 -> n=lane-16, elems e = K 8..15, 24..31
//   Global K ordering: k = tap*32 + cin  (one K-step == one tap, all channels)
// ---------------------------------------------------------------------------
__global__ __launch_bounds__(256) void k_pack_w(const float* __restrict__ w_off,
                                                const float* __restrict__ w_mask,
                                                const float* __restrict__ w,
                                                _Float16* __restrict__ w1f,
                                                _Float16* __restrict__ w2f) {
    int f = blockIdx.x * 256 + threadIdx.x;
    if (f < W1F_ELEMS) {
        int e    = f & 15;
        int lane = (f >> 4) & 31;
        int nt   = (f >> 9) % 7;
        int s    = f / 3584;                 // tap
        int n    = lane & 15;
        int cin  = ((e >> 3) << 4) + ((lane >> 4) << 3) + (e & 7);   // k_local
        int chan = nt * 16 + n;
        float v = 0.f;
        if (chan < 81)       v = w_off[(chan * CIN_ + cin) * KTAPS + s];
        else if (chan < 108) v = w_mask[((chan - 81) * CIN_ + cin) * KTAPS + s];
        w1f[f] = (_Float16)v;
    } else if (f < W1F_ELEMS + W2F_ELEMS) {
        int g    = f - W1F_ELEMS;
        int e    = g & 15;
        int lane = (g >> 4) & 31;
        int nt2  = (g >> 9) & 1;
        int s    = g / 1024;                 // tap
        int cout = nt2 * 16 + (lane & 15);
        int cin  = ((e >> 3) << 4) + ((lane >> 4) << 3) + (e & 7);
        w2f[g] = (_Float16)w[(cout * CIN_ + cin) * KTAPS + s];
    }
}

__device__ __forceinline__ int tap_sidx(int s, int z0, int y0, int xcol, int hh) {
    const int kz = s / 9, ky = (s % 9) / 3, kx = s % 3;
    return (((z0 + kz) * PH + (y0 + ky)) * PW + (xcol + kx)) * CIN_ + hh * 8;
}

// ---------------------------------------------------------------------------
// Kernel 3 (fused): per wave32, one M-tile of 16 spatial positions.
//  Phase A: offset/mask implicit GEMM (M16 x N112 x K864), 27 K-steps x 7 WMMA.
//           BOTH A and B fragments register-double-buffered one K-step ahead
//           so the 7-WMMA burst hides all fragment load latency; results
//           transposed through LDS (16 rows x 112 chans, f32).
//  Phase B: deformable trilinear gather accumulated in packed f16
//           (v_pk_fma_f16), mask applied in f16, 27 x 2 WMMA vs packed w
//           (B frags double-buffered), + bias, store NCDHW f32.
// ---------------------------------------------------------------------------
__global__ __launch_bounds__(256) void k_fused(const _Float16* __restrict__ xpad,
                                               const _Float16* __restrict__ w1f,
                                               const _Float16* __restrict__ w2f,
                                               const float* __restrict__ b_off,
                                               const float* __restrict__ b_mask,
                                               const float* __restrict__ bias,
                                               float* __restrict__ out) {
    __shared__ float lds[8 * 16 * 112];      // 57,344 B per 8-wave block
    const int lane = threadIdx.x & 31;
    const int wave = threadIdx.x >> 5;
    const int mtile = blockIdx.x * 8 + wave;
    const int posbase = mtile * 16;
    const int x0 = posbase & 63;
    const int y0 = (posbase >> 6) & 63;
    const int z0 = posbase >> 12;
    const int ln = lane & 15;                // A: M-row; C/D: N-col
    const int hh = lane >> 4;                // lane half
    const int xcol = x0 + ln;                // tile always within one x-row (64%16==0)
    float* my = lds + wave * 16 * 112;

    // ---------------- Phase A: offset+mask GEMM (fully pipelined) ----------
    v8f acc[7];
    #pragma unroll
    for (int nt = 0; nt < 7; ++nt)
        #pragma unroll
        for (int i = 0; i < 8; ++i) acc[nt][i] = 0.f;

    // preload step 0: A-fragment halves and all 7 B fragments
    int sidx = tap_sidx(0, z0, y0, xcol, hh);
    v8h g0 = *(const v8h*)(xpad + sidx);
    v8h g1 = *(const v8h*)(xpad + sidx + 16);
    v16h bfr[7];
    {
        const _Float16* wb0 = w1f + lane * 16;
        #pragma unroll
        for (int nt = 0; nt < 7; ++nt) bfr[nt] = *(const v16h*)(wb0 + nt * 512);
    }

    for (int s = 0; s < 27; ++s) {
        const v16h a = __builtin_shufflevector(g0, g1,
            0, 1, 2, 3, 4, 5, 6, 7, 8, 9, 10, 11, 12, 13, 14, 15);
        v16h bnx[7];
        if (s < 26) {   // issue next step's A and B loads before the WMMA burst
            const int nidx = tap_sidx(s + 1, z0, y0, xcol, hh);
            g0 = *(const v8h*)(xpad + nidx);
            g1 = *(const v8h*)(xpad + nidx + 16);
            const _Float16* wbn = w1f + (s + 1) * 3584 + lane * 16;
            __builtin_prefetch(wbn + 3584, 0, 1);   // s+2 B frags into cache
            #pragma unroll
            for (int nt = 0; nt < 7; ++nt) bnx[nt] = *(const v16h*)(wbn + nt * 512);
        }
        #pragma unroll
        for (int nt = 0; nt < 7; ++nt)
            acc[nt] = __builtin_amdgcn_wmma_f32_16x16x32_f16(
                false, a, false, bfr[nt], (short)0, acc[nt], false, false);
        if (s < 26) {
            #pragma unroll
            for (int nt = 0; nt < 7; ++nt) bfr[nt] = bnx[nt];
        }
    }
    // C layout: VGPR i -> row M=i(+8 for lanes 16..31), N = lane&15
    #pragma unroll
    for (int nt = 0; nt < 7; ++nt)
        #pragma unroll
        for (int i = 0; i < 8; ++i)
            my[(i + hh * 8) * 112 + nt * 16 + ln] = acc[nt][i];
    __syncthreads();

    // ---------------- Phase B: deformable gather + GEMM ----------------
    v8f oa0, oa1;
    #pragma unroll
    for (int i = 0; i < 8; ++i) { oa0[i] = 0.f; oa1[i] = 0.f; }

    // preload tap 0 B fragments
    v16h b0 = *(const v16h*)(w2f + lane * 16);
    v16h b1 = *(const v16h*)(w2f + lane * 16 + 512);

    const float* r = my + ln * 112;          // this lane's position row
    for (int tap = 0; tap < 27; ++tap) {
        const float oz = r[tap]        + b_off[tap];
        const float oy = r[27 + tap]   + b_off[27 + tap];
        const float ox = r[54 + tap]   + b_off[54 + tap];
        const float m  = 1.f / (1.f + __expf(-(r[81 + tap] + b_mask[tap])));
        const int kz = tap / 9, ky = (tap % 9) / 3, kx = tap % 3;
        const float cz = (float)(z0 + kz - 1) + oz;
        const float cy = (float)(y0 + ky - 1) + oy;
        const float cx = (float)(xcol + kx - 1) + ox;
        const float fz = floorf(cz), fy = floorf(cy), fx = floorf(cx);
        const int iz = (int)fz, iy = (int)fy, ix = (int)fx;
        const float tz = cz - fz, ty = cy - fy, tx = cx - fx;

        // next tap's B fragments issued before the gather+WMMA work
        v16h b0n, b1n;
        if (tap < 26) {
            const _Float16* wbn2 = w2f + (tap + 1) * 1024 + lane * 16;
            __builtin_prefetch(wbn2 + 1024, 0, 1);
            b0n = *(const v16h*)(wbn2);
            b1n = *(const v16h*)(wbn2 + 512);
        }

        // packed-f16 trilinear accumulation (v_pk_fma_f16)
        v8h ah0, ah1;
        #pragma unroll
        for (int j = 0; j < 8; ++j) { ah0[j] = (_Float16)0.f; ah1[j] = (_Float16)0.f; }

        #pragma unroll
        for (int c = 0; c < 8; ++c) {
            const int dz = c >> 2, dy = (c >> 1) & 1, dx = c & 1;
            const int zi = iz + dz, yi = iy + dy, xi = ix + dx;
            const bool ok = (zi >= 0) && (zi < DD) && (yi >= 0) && (yi < HH) &&
                            (xi >= 0) && (xi < WW);
            const float wgt = (dz ? tz : 1.f - tz) * (dy ? ty : 1.f - ty) *
                              (dx ? tx : 1.f - tx) * (ok ? 1.f : 0.f);
            const int zc = min(max(zi, 0), DD - 1);
            const int yc = min(max(yi, 0), HH - 1);
            const int xc = min(max(xi, 0), WW - 1);
            const int gi = (((zc + 1) * PH + (yc + 1)) * PW + (xc + 1)) * CIN_ + hh * 8;
            const v8h g0b = *(const v8h*)(xpad + gi);
            const v8h g1b = *(const v8h*)(xpad + gi + 16);
            const _Float16 wh = (_Float16)wgt;
            ah0 += g0b * wh;    // 4x v_pk_fma_f16
            ah1 += g1b * wh;    // 4x v_pk_fma_f16
        }
        const _Float16 mh = (_Float16)m;
        ah0 *= mh;
        ah1 *= mh;
        const v16h a = __builtin_shufflevector(ah0, ah1,
            0, 1, 2, 3, 4, 5, 6, 7, 8, 9, 10, 11, 12, 13, 14, 15);

        oa0 = __builtin_amdgcn_wmma_f32_16x16x32_f16(false, a, false, b0, (short)0, oa0, false, false);
        oa1 = __builtin_amdgcn_wmma_f32_16x16x32_f16(false, a, false, b1, (short)0, oa1, false, false);
        if (tap < 26) { b0 = b0n; b1 = b1n; }
    }

    // store: D layout -> out[(cout)*NSPATIAL + pos], NCDHW
    #pragma unroll
    for (int i = 0; i < 8; ++i) {
        const int pos = posbase + i + hh * 8;
        out[ln * NSPATIAL + pos]        = oa0[i] + bias[ln];
        out[(16 + ln) * NSPATIAL + pos] = oa1[i] + bias[16 + ln];
    }
}

// ---------------------------------------------------------------------------
extern "C" void kernel_launch(void* const* d_in, const int* in_sizes, int n_in,
                              void* d_out, int out_size, void* d_ws, size_t ws_size,
                              hipStream_t stream) {
    const float* x      = (const float*)d_in[0];
    const float* w_off  = (const float*)d_in[1];
    const float* b_off  = (const float*)d_in[2];
    const float* w_mask = (const float*)d_in[3];
    const float* b_mask = (const float*)d_in[4];
    const float* w      = (const float*)d_in[5];
    const float* b      = (const float*)d_in[6];

    _Float16* xpad = (_Float16*)d_ws;
    _Float16* w1f  = (_Float16*)((char*)d_ws + (size_t)XPAD_ELEMS * sizeof(_Float16));
    _Float16* w2f  = w1f + W1F_ELEMS;

    k_pad_x<<<(XPAD_ELEMS + 255) / 256, 256, 0, stream>>>(x, xpad);
    k_pack_w<<<(W1F_ELEMS + W2F_ELEMS + 255) / 256, 256, 0, stream>>>(w_off, w_mask, w, w1f, w2f);
    // 8192 M-tiles of 16 positions, 8 wave32 waves per 256-thread block
    k_fused<<<1024, 256, 0, stream>>>(xpad, w1f, w2f, b_off, b_mask, b, (float*)d_out);
}
